// RNN_33200097198343
// MI455X (gfx1250) — matrix operand order repaired and checked by reading, not compiled, not run on hardware
//
#include <hip/hip_runtime.h>
#include <hip/hip_bf16.h>
#include <math.h>

// Problem sizes (fixed by the reference): x(64,512,1024), W(2048,1024), b(1,1024)
#define N_B   64
#define T_S   512
#define F_D   1024
#define H_D   1024
#define ROWS  (N_B * T_S)   // 32768 rows of the big GEMM (n*T + t indexing)
#define G2    8             // persistent blocks in the recurrent phase

// Phase-1 LDS-tiled GEMM geometry
#define BM 128
#define BN 128
#define BK 64

typedef __attribute__((ext_vector_type(16))) __bf16 v16bf;
typedef __attribute__((ext_vector_type(8)))  float  v8f;

union BFrag { v16bf v; uint4 q[2]; };

static __device__ inline unsigned short f2bf_bits(float f) {
  union { float f; unsigned u; } c; c.f = f;
  unsigned r = c.u + 0x7FFFu + ((c.u >> 16) & 1u);   // round to nearest even
  return (unsigned short)(r >> 16);
}

static __device__ inline __bf16 f2bf(float f) {
  union { unsigned short u; __bf16 b; } o;
  o.u = f2bf_bits(f);
  return o.b;
}

// A fragment (16x32, bf16) from f32 memory with on-the-fly conversion
// (fallback path only). p = &A[row = lane&15][kb + (lane>>4)*8].
static __device__ inline v16bf load_a_from_f32(const float* __restrict__ p) {
  const float4* q0 = (const float4*)p;
  const float4* q1 = (const float4*)(p + 16);
  float4 a0 = q0[0], a1 = q0[1], b0 = q1[0], b1 = q1[1];
  v16bf v;
  v[0]=f2bf(a0.x);  v[1]=f2bf(a0.y);  v[2]=f2bf(a0.z);  v[3]=f2bf(a0.w);
  v[4]=f2bf(a1.x);  v[5]=f2bf(a1.y);  v[6]=f2bf(a1.z);  v[7]=f2bf(a1.w);
  v[8]=f2bf(b0.x);  v[9]=f2bf(b0.y);  v[10]=f2bf(b0.z); v[11]=f2bf(b0.w);
  v[12]=f2bf(b1.x); v[13]=f2bf(b1.y); v[14]=f2bf(b1.z); v[15]=f2bf(b1.w);
  return v;
}

// A fragment (16x32, bf16): lane holds K=[k0..k0+7] and [k0+16..k0+23].
static __device__ inline v16bf load_a_from_bf16(const __bf16* p) {
  BFrag f;
  f.q[0] = *(const uint4*)p;
  f.q[1] = *(const uint4*)(p + 16);
  return f.v;
}

// B fragment (32x16, bf16) from W^T layout: lane holds K=[k0..k0+7] and [k0+16..k0+23] of one column.
static __device__ inline v16bf load_b_from_bf16(const __bf16* p) {
  BFrag f;
  f.q[0] = *(const uint4*)p;
  f.q[1] = *(const uint4*)(p + 16);
  return f.v;
}

static __device__ inline v8f wmma_bf16(v16bf a, v16bf b, v8f c) {
  return __builtin_amdgcn_wmma_f32_16x16x32_bf16(false, a, false, b, (short)0, c,
                                                 false, false);
}

// ---------------------------------------------------------------------------
// Prep: W (2048x1024 f32) -> WxT[1024][1024] bf16, WhT[1024][1024] bf16 (transposed).
__global__ void __launch_bounds__(256)
prep_w_kernel(const float* __restrict__ W, __bf16* __restrict__ WxT,
              __bf16* __restrict__ WhT) {
  int idx = blockIdx.x * 256 + threadIdx.x;   // 0 .. 2*1024*1024-1
  int k = idx >> 10;
  int j = idx & 1023;
  __bf16 bw = f2bf(W[idx]);
  if (k < F_D) WxT[(size_t)j * F_D + k] = bw;
  else         WhT[(size_t)j * H_D + (k - F_D)] = bw;
}

__global__ void __launch_bounds__(256)
prep_state_kernel(unsigned* __restrict__ p, int n) {
  int idx = blockIdx.x * 256 + threadIdx.x;
  if (idx < n) p[idx] = 0u;
}

// One-shot x f32 -> bf16 conversion (4 elements / thread, streaming).
__global__ void __launch_bounds__(256)
conv_x_kernel(const float* __restrict__ x, uint2* __restrict__ xb) {
  size_t i = (size_t)blockIdx.x * 256 + threadIdx.x;
  float4 v = ((const float4*)x)[i];
  uint2 o;
  o.x = (unsigned)f2bf_bits(v.x) | ((unsigned)f2bf_bits(v.y) << 16);
  o.y = (unsigned)f2bf_bits(v.z) | ((unsigned)f2bf_bits(v.w) << 16);
  xb[i] = o;
}

// ---------------------------------------------------------------------------
// Phase 1 (main path): Xp = x @ Wx + b with LDS tiling.
// Block tile 128M x 128N, 8 waves as 4(M) x 2(N), wave tile 32x64, K-chunk 64.
__global__ void __launch_bounds__(256)
xw_gemm_lds_kernel(const __bf16* __restrict__ xbf, const __bf16* __restrict__ WxT,
                   const float* __restrict__ bias, float* __restrict__ z) {
  __shared__ __bf16 As[BM * BK];   // 16 KB  [row][k]
  __shared__ __bf16 Bs[BN * BK];   // 16 KB  [col][k]

  const int lane = threadIdx.x & 31;
  const int wid  = threadIdx.x >> 5;
  const int wm   = wid >> 1;                    // 0..3
  const int wn   = wid & 1;                     // 0..1
  const int mbase = (blockIdx.x >> 3) * BM;     // 256 M-tiles
  const int nbase = (blockIdx.x & 7) * BN;      // 8 N-tiles
  const int r    = lane & 15;
  const int half = lane >> 4;

  v8f acc[2][4] = {};

  for (int kc = 0; kc < F_D; kc += BK) {
    __syncthreads();
    // Cooperative stage: A (1024 uint4) + B (1024 uint4), 4+4 per thread.
#pragma unroll
    for (int i = 0; i < 4; ++i) {
      int idx = threadIdx.x + i * 256;          // 0..1023
      int row = idx >> 3, off = (idx & 7) * 8;
      ((uint4*)As)[idx] =
          *(const uint4*)(xbf + (size_t)(mbase + row) * F_D + kc + off);
      ((uint4*)Bs)[idx] =
          *(const uint4*)(WxT + (size_t)(nbase + row) * F_D + kc + off);
    }
    __syncthreads();

#pragma unroll
    for (int kk = 0; kk < BK; kk += 32) {
      v16bf fa0 = load_a_from_bf16(As + (wm * 32 + r)      * BK + kk + half * 8);
      v16bf fa1 = load_a_from_bf16(As + (wm * 32 + 16 + r) * BK + kk + half * 8);
#pragma unroll
      for (int j = 0; j < 4; ++j) {
        v16bf fb = load_b_from_bf16(Bs + (wn * 64 + j * 16 + r) * BK + kk + half * 8);
        acc[0][j] = wmma_bf16(fa0, fb, acc[0][j]);
        acc[1][j] = wmma_bf16(fa1, fb, acc[1][j]);
      }
    }
  }

  const int mrow = mbase + wm * 32;
  const int nb   = nbase + wn * 64;
#pragma unroll
  for (int j = 0; j < 4; ++j) {
    const int col = nb + j * 16 + r;
    const float bv = bias[col];
#pragma unroll
    for (int i = 0; i < 2; ++i) {
#pragma unroll
      for (int rr = 0; rr < 8; ++rr) {
        const int row = mrow + i * 16 + rr + half * 8;
        z[(size_t)row * H_D + col] = acc[i][j][rr] + bv;
      }
    }
  }
}

// Phase 1 (fallback if workspace can't hold bf16 x): inline f32->bf16 conversion.
__global__ void __launch_bounds__(256)
xw_gemm_kernel(const float* __restrict__ x, const __bf16* __restrict__ WxT,
               const float* __restrict__ bias, float* __restrict__ z) {
  const int lane = threadIdx.x & 31;
  const int wid  = threadIdx.x >> 5;
  const int wg   = blockIdx.x * 8 + wid;
  const int mrow = (wg >> 4) * 32;
  const int nb   = (wg & 15) * 64;
  const int r    = lane & 15;
  const int half = lane >> 4;

  v8f acc[2][4] = {};
  const float* a0 = x + (size_t)(mrow + r) * F_D + half * 8;
  const float* a1 = a0 + (size_t)16 * F_D;

  for (int kb = 0; kb < F_D; kb += 32) {
    __builtin_prefetch(a0 + kb + 128, 0, 1);
    v16bf fa0 = load_a_from_f32(a0 + kb);
    v16bf fa1 = load_a_from_f32(a1 + kb);
#pragma unroll
    for (int j = 0; j < 4; ++j) {
      const __bf16* bp = WxT + (size_t)(nb + j * 16 + r) * F_D + kb + half * 8;
      v16bf fb = load_b_from_bf16(bp);
      acc[0][j] = wmma_bf16(fa0, fb, acc[0][j]);
      acc[1][j] = wmma_bf16(fa1, fb, acc[1][j]);
    }
  }
#pragma unroll
  for (int j = 0; j < 4; ++j) {
    const int col = nb + j * 16 + r;
    const float bv = bias[col];
#pragma unroll
    for (int i = 0; i < 2; ++i) {
#pragma unroll
      for (int rr = 0; rr < 8; ++rr) {
        const int row = mrow + i * 16 + rr + half * 8;
        z[(size_t)row * H_D + col] = acc[i][j][rr] + bv;
      }
    }
  }
}

// ---------------------------------------------------------------------------
// Phase 2: persistent recurrent kernel (8 blocks x 8 waves; wave w owns h
// columns [16w,16w+16)). Per step: stage h_{t-1} to LDS, h@Wh via WMMA,
// add Xp (in z), tanh, write z + bf16 h broadcast, grid barrier.
__global__ void __launch_bounds__(256)
rnn_steps_kernel(const __bf16* __restrict__ WhT, float* __restrict__ z,
                 __bf16* __restrict__ hbuf, unsigned* __restrict__ cnt,
                 float* __restrict__ hlast) {
  extern __shared__ unsigned char smem[];
  __bf16* hs = (__bf16*)smem;                 // 64 x 1024 bf16 = 128 KB

  const int lane = threadIdx.x & 31;
  const int wid  = threadIdx.x >> 5;
  const int w    = blockIdx.x * 8 + wid;      // 0..63
  const int nb   = w * 16;
  const int r    = lane & 15;
  const int half = lane >> 4;
  const int col  = nb + r;

  const __bf16* bBase = WhT + (size_t)col * H_D + half * 8;
  const __bf16* aBase = hs + (size_t)r * H_D + half * 8;

  for (int t = 0; t < T_S; ++t) {
    const __bf16* hprev = hbuf + (size_t)(t & 1) * (N_B * H_D);
    __bf16* hnext       = hbuf + (size_t)((t + 1) & 1) * (N_B * H_D);

    {
      const uint4* src = (const uint4*)hprev;
      uint4* dst = (uint4*)hs;
#pragma unroll
      for (int i = 0; i < 32; ++i)
        dst[threadIdx.x + i * 256] = src[threadIdx.x + i * 256];
    }
    __syncthreads();

    v8f acc[4] = {};
    for (int kb = 0; kb < H_D; kb += 32) {
      v16bf fb = load_b_from_bf16(bBase + kb);
#pragma unroll
      for (int i = 0; i < 4; ++i) {
        v16bf fa = load_a_from_bf16(aBase + (size_t)(i * 16) * H_D + kb);
        acc[i] = wmma_bf16(fa, fb, acc[i]);
      }
    }

#pragma unroll
    for (int i = 0; i < 4; ++i) {
#pragma unroll
      for (int rr = 0; rr < 8; ++rr) {
        const int n = i * 16 + rr + half * 8;
        const size_t zi = ((size_t)n * T_S + t) * H_D + col;
        const float hv = tanhf(acc[i][rr] + z[zi]);        // z holds Xp(+b)
        z[zi] = hv;
        hnext[(size_t)n * H_D + col] = f2bf(hv);
        if (t == T_S - 1) hlast[(size_t)n * H_D + col] = hv;
      }
    }

    __threadfence();
    __syncthreads();
    if (threadIdx.x == 0) {
      (void)atomicAdd(&cnt[t], 1u);
      while (__hip_atomic_load(&cnt[t], __ATOMIC_RELAXED,
                               __HIP_MEMORY_SCOPE_AGENT) < G2) {
        __builtin_amdgcn_s_sleep(1);
      }
    }
    __syncthreads();
    __threadfence();
  }
}

// ---------------------------------------------------------------------------
extern "C" void kernel_launch(void* const* d_in, const int* in_sizes, int n_in,
                              void* d_out, int out_size, void* d_ws, size_t ws_size,
                              hipStream_t stream) {
  const float* x = (const float*)d_in[0];    // (64,512,1024) f32
  const float* W = (const float*)d_in[1];    // (2048,1024) f32
  const float* b = (const float*)d_in[2];    // (1,1024) f32

  float* z     = (float*)d_out;              // (N,T,H) f32; doubles as Xp buffer
  float* hlast = z + (size_t)ROWS * H_D;

  // Workspace layout:
  //   [0, 2MB)              WxT  bf16 [1024][1024]
  //   [2MB, 4MB)            WhT  bf16 [1024][1024]
  //   [4MB, +256KB)         hbuf bf16 [2][64][1024]
  //   [.., +2KB)            cnt  u32  [512]
  //   [4.5MB, +64MB)        xbf  bf16 [32768][1024]   (main path only)
  char* ws = (char*)d_ws;
  __bf16*  WxT  = (__bf16*)(ws);
  __bf16*  WhT  = (__bf16*)(ws + (size_t)(1 << 21));
  __bf16*  hbuf = (__bf16*)(ws + (size_t)(2 << 21));
  unsigned* cnt = (unsigned*)(ws + (size_t)(2 << 21) +
                              (size_t)2 * N_B * H_D * sizeof(__bf16));
  const size_t xbfOff   = (size_t)4608 * 1024;           // 4.5 MB, aligned
  __bf16*  xbf  = (__bf16*)(ws + xbfOff);
  const size_t xbfBytes = (size_t)ROWS * F_D * 2;        // 64 MB
  const bool bigWs = ws_size >= xbfOff + xbfBytes;

  prep_w_kernel<<<(2 * F_D * H_D) / 256, 256, 0, stream>>>(W, WxT, WhT);
  const int stateWords = (2 * N_B * H_D * 2 + T_S * 4) / 4;
  prep_state_kernel<<<(stateWords + 255) / 256, 256, 0, stream>>>(
      (unsigned*)(ws + (size_t)(2 << 21)), stateWords);

  if (bigWs) {
    conv_x_kernel<<<(int)((size_t)ROWS * F_D / 4 / 256), 256, 0, stream>>>(
        x, (uint2*)xbf);
    xw_gemm_lds_kernel<<<(ROWS / BM) * (H_D / BN), 256, 0, stream>>>(
        xbf, WxT, b, z);
  } else {
    xw_gemm_kernel<<<(ROWS / 32) * (H_D / 64) / 8, 256, 0, stream>>>(x, WxT, b, z);
  }

  rnn_steps_kernel<<<G2, 256, N_B * H_D * sizeof(__bf16), stream>>>(
      WhT, z, hbuf, cnt, hlast);
}
